// SpMiddleResNetFHD_27839978012927
// MI455X (gfx1250) — matrix operand order repaired and checked by reading, not compiled, not run on hardware
//
#include <hip/hip_runtime.h>
#include <hip/hip_bf16.h>
#include <stdint.h>

// ---------------------------------------------------------------------------
// SpMiddleResNetFHD on MI455X (gfx1250, wave32, WMMA).
// Every conv is an implicit GEMM on v_wmma_f32_16x16x32_f16:
//   M = output voxels (tile of 128 per block), N = Cout, K = Cin per 3D tap.
// Activations are stored dense f16 [D][H][W][C]; BN folded to scale/shift;
// bias / residual / ReLU / sparsity-mask fused into the epilogue.
// A-tile staging uses CDNA5 async global->LDS (ASYNCcnt) for in-bounds rows.
// ---------------------------------------------------------------------------

typedef __attribute__((ext_vector_type(16))) _Float16 v16h;
typedef __attribute__((ext_vector_type(8)))  float    v8f;

#define BN_EPS 1e-3f

struct ConvDims {
  int Din, Hin, Win;
  int Dout, Hout, Wout;
  int Cin, CinPad, Cout;
  int kd, kh, kw;
  int sd, sh, sw;
  int pd, ph, pw;
  int relu, addid;
  int nvoxOut, HWout;
};

// ---------------------------------------------------------------------------
// Generic WMMA implicit-GEMM conv.  Block = 256 threads = 8 wave32.
// Each block: 128 output voxels x Cout.  Wave w owns voxel rows w*16..w*16+15.
// NT = Cout/16 accumulator tiles held in registers (templated: constant idx).
// LDS tiles use an 80-byte row pitch (40 halfs) -> conflict-free b128 reads.
// ---------------------------------------------------------------------------
template <int NT>
__global__ __launch_bounds__(256)
void conv_wmma_kernel(const _Float16* __restrict__ xin,
                      const _Float16* __restrict__ wts,   // [T][CinPad][Cout] f16
                      const float*    __restrict__ scsh,  // [Cout] scale, [Cout] shift
                      const unsigned char* __restrict__ maskOut,
                      const _Float16* __restrict__ ident, // residual input (or null)
                      _Float16* __restrict__ xout,
                      ConvDims cd)
{
  __shared__ __align__(16) _Float16 As[128 * 40];
  __shared__ __align__(16) _Float16 Bs[128 * 40];

  const int tid   = threadIdx.x;
  const int lane  = tid & 31;
  const int wv    = tid >> 5;       // wave id 0..7
  const int asel  = lane >> 4;      // 0: lanes 0-15, 1: lanes 16-31
  const int lrow  = lane & 15;
  const int tile0 = blockIdx.x * 128;

  const int T       = cd.kd * cd.kh * cd.kw;
  const int cchunks = (cd.CinPad + 31) >> 5;

  v8f acc[NT];
  {
    v8f z = {};
#pragma unroll
    for (int i = 0; i < NT; ++i) acc[i] = z;
  }

  // This thread's A-load assignment: row = tid/2, half-slots (tid&1)*16..+15.
  const int arow_ld = tid >> 1;
  const int ahalf   = (tid & 1) * 16;
  const int voxel_ld = tile0 + arow_ld;
  const bool vvalid  = voxel_ld < cd.nvoxOut;
  int od = 0, oh = 0, ow = 0;
  if (vvalid) {
    od = voxel_ld / cd.HWout;
    int r = voxel_ld - od * cd.HWout;
    oh = r / cd.Wout;
    ow = r - oh * cd.Wout;
  }
  const int bid = od * cd.sd - cd.pd;
  const int bih = oh * cd.sh - cd.ph;
  const int biw = ow * cd.sw - cd.pw;

  // LDS byte address of this thread's A slot (flat shared ptr low 32 bits ==
  // LDS byte address per the gfx1250 aperture rule).
  _Float16* adst = &As[arow_ld * 40 + ahalf];
  const uint32_t adst_lds = (uint32_t)(uintptr_t)adst;

  for (int t = 0; t < T; ++t) {
    const int tz = t / (cd.kh * cd.kw);
    const int tr = t - tz * (cd.kh * cd.kw);
    const int ty = tr / cd.kw;
    const int tx = tr - ty * cd.kw;
    const int di = bid + tz, hi = bih + ty, wi = biw + tx;
    const bool inb = vvalid && di >= 0 && di < cd.Din && hi >= 0 && hi < cd.Hin &&
                     wi >= 0 && wi < cd.Win;
    const _Float16* srcRow =
        xin + ((size_t)((size_t)di * cd.Hin + hi) * cd.Win + wi) * cd.CinPad;

    // Prefetch next tap's weight panel into cache (global_prefetch_b8 path).
    if (t + 1 < T)
      __builtin_prefetch(wts + (size_t)(t + 1) * cd.CinPad * cd.Cout, 0, 1);

    for (int cc = 0; cc < cchunks; ++cc) {
      const int c0 = cc << 5;
      int ccount = cd.CinPad - c0; if (ccount > 32) ccount = 32;

      // ---- A tile: 128 rows x 32 halfs (gather, zero-pad OOB / tail) ----
      {
        const int avail = ccount - ahalf;
        if (inb && avail >= 16) {
          // CDNA5 async global->LDS copy: 2 x b128, tracked by ASYNCcnt.
          const uint64_t g0 = (uint64_t)(uintptr_t)(srcRow + c0 + ahalf);
          const uint64_t g1 = g0 + 16u;
          asm volatile(
              "global_load_async_to_lds_b128 %0, %1, off\n\t"
              "global_load_async_to_lds_b128 %2, %3, off"
              :
              : "v"(adst_lds), "v"(g0), "v"(adst_lds + 16u), "v"(g1)
              : "memory");
        } else {
#pragma unroll
          for (int k = 0; k < 16; ++k) {
            const int kk = ahalf + k;
            _Float16 v = (_Float16)0.f;
            if (inb && kk < ccount) v = srcRow[c0 + kk];
            adst[k] = v;
          }
        }
      }

      // ---- B tile: Cout rows x 32 halfs, Bs[n][k] = W[t][c0+k][n] ----
      for (int it = tid; it < cd.Cout * 2; it += 256) {
        const int n    = it >> 1;
        const int half = (it & 1) * 16;
        _Float16* dst  = &Bs[n * 40 + half];
        const _Float16* wsrc =
            wts + ((size_t)t * cd.CinPad + c0 + half) * cd.Cout + n;
#pragma unroll
        for (int k = 0; k < 16; ++k) {
          const int kk = half + k;
          dst[k] = (kk < ccount) ? wsrc[(size_t)k * cd.Cout] : (_Float16)0.f;
        }
      }
      // Drain async LDS writes, then make the tile visible to all waves.
      asm volatile("s_wait_asynccnt 0x0" ::: "memory");
      __syncthreads();

      // ---- fragments per ISA lane layout, then WMMA ----
      union Frag { uint4 u[2]; v16h h; };
      Frag af;
      {
        // A 16x32 f16: lanes 0-15 hold K0..7 & K16..23; lanes 16-31 K8..15 & K24..31
        const uint4* Arow = (const uint4*)&As[(wv * 16 + lrow) * 40];
        af.u[0] = Arow[asel];
        af.u[1] = Arow[2 + asel];
      }
#pragma unroll
      for (int nc = 0; nc < NT; ++nc) {
        // B 32x16 f16: lanes 0-15 hold K0..7 & K16..23 (col N=lane); lanes 16-31 K8..15 & K24..31
        Frag bf;
        const uint4* Brow = (const uint4*)&Bs[(nc * 16 + lrow) * 40];
        bf.u[0] = Brow[asel];
        bf.u[1] = Brow[2 + asel];
        acc[nc] = __builtin_amdgcn_wmma_f32_16x16x32_f16(
            false, af.h, false, bf.h, (short)0, acc[nc], false, false);
      }
      __syncthreads();
    }
  }

  // ---- epilogue: BN scale/shift (+bias folded), residual, ReLU, mask gate ----
  float sc[NT], sh[NT];
#pragma unroll
  for (int nc = 0; nc < NT; ++nc) {
    const int ch = nc * 16 + lrow;
    sc[nc] = scsh[ch];
    sh[nc] = scsh[cd.Cout + ch];
  }
#pragma unroll
  for (int r = 0; r < 8; ++r) {
    const int mrow  = wv * 16 + (asel ? (8 + r) : r); // C/D layout: lanes>=16 -> M=8+r
    const int voxel = tile0 + mrow;
    if (voxel >= cd.nvoxOut) continue;
    const bool on = maskOut[voxel] != 0;
#pragma unroll
    for (int nc = 0; nc < NT; ++nc) {
      const int ch = nc * 16 + lrow;
      float val = acc[nc][r] * sc[nc] + sh[nc];
      if (cd.addid) val += (float)ident[(size_t)voxel * cd.Cout + ch];
      if (cd.relu)  val = fmaxf(val, 0.f);
      if (!on)      val = 0.f;
      xout[(size_t)voxel * cd.Cout + ch] = (_Float16)val;
    }
  }
}

// ---------------------------------------------------------------------------
// Helper kernels
// ---------------------------------------------------------------------------
__global__ void zero_words_kernel(uint32_t* p, long n) {
  long i = (long)blockIdx.x * blockDim.x + threadIdx.x;
  long stride = (long)gridDim.x * blockDim.x;
  for (; i < n; i += stride) p[i] = 0u;
}

__global__ void scatter_kernel(const float* __restrict__ vox,
                               const int* __restrict__ coors, int n,
                               _Float16* __restrict__ x,
                               unsigned char* __restrict__ mask,
                               int H, int W, int Cpad) {
  int i = blockIdx.x * blockDim.x + threadIdx.x;
  if (i >= n) return;
  const int z = coors[i * 4 + 1], y = coors[i * 4 + 2], xw = coors[i * 4 + 3];
  const long v = ((long)z * H + y) * W + xw;
  mask[v] = 1;
  _Float16* d = x + v * Cpad;
#pragma unroll
  for (int c = 0; c < 5; ++c) d[c] = (_Float16)vox[i * 5 + c];
}

// OIDHW f32 -> [tap][CinPad][Cout] f16 (zero-padded input channels)
__global__ void prep_w_kernel(const float* __restrict__ w, _Float16* __restrict__ dst,
                              int Cout, int Cin, int CinPad, int kd, int kh, int kw) {
  const int total = kd * kh * kw * CinPad * Cout;
  const int idx = blockIdx.x * blockDim.x + threadIdx.x;
  if (idx >= total) return;
  const int co = idx % Cout;
  int rest = idx / Cout;
  const int ci = rest % CinPad;
  const int t  = rest / CinPad;
  const int x = t % kw, y = (t / kw) % kh, z = t / (kw * kh);
  float val = 0.f;
  if (ci < Cin)
    val = w[((((long)co * Cin + ci) * kd + z) * kh + y) * kw + x];
  dst[idx] = (_Float16)val;
}

// Fold eval-mode BN (+ optional conv bias) into per-channel scale/shift.
__global__ void prep_scsh_kernel(const float* g, const float* b, const float* m,
                                 const float* v, const float* bias, int C,
                                 float* scsh) {
  const int c = blockIdx.x * blockDim.x + threadIdx.x;
  if (c >= C) return;
  const float sc = g[c] * rsqrtf(v[c] + BN_EPS);
  float sh = b[c] - m[c] * sc;
  if (bias) sh += bias[c] * sc;
  scsh[c] = sc;
  scsh[C + c] = sh;
}

// SparseConv active-set dilation: new_mask = (conv(mask, ones) > 0)
__global__ void mask_down_kernel(const unsigned char* __restrict__ mi,
                                 unsigned char* __restrict__ mo,
                                 int Din, int Hin, int Win,
                                 int Dout, int Hout, int Wout,
                                 int kd, int kh, int kw,
                                 int sd, int sh, int sw,
                                 int pd, int ph, int pw) {
  const int n = Dout * Hout * Wout;
  const int idx = blockIdx.x * blockDim.x + threadIdx.x;
  if (idx >= n) return;
  const int d = idx / (Hout * Wout);
  int r = idx - d * Hout * Wout;
  const int h = r / Wout, w = r - (r / Wout) * Wout;
  unsigned char on = 0;
  for (int z = 0; z < kd && !on; ++z) {
    const int di = d * sd - pd + z;
    if (di < 0 || di >= Din) continue;
    for (int y = 0; y < kh && !on; ++y) {
      const int hi = h * sh - ph + y;
      if (hi < 0 || hi >= Hin) continue;
      for (int x = 0; x < kw; ++x) {
        const int wi = w * sw - pw + x;
        if (wi < 0 || wi >= Win) continue;
        if (mi[((long)di * Hin + hi) * Win + wi]) { on = 1; break; }
      }
    }
  }
  mo[idx] = on;
}

// x5 [2][40][40][128] f16  ->  out [256][40][40] f32, channel = c*2 + d
__global__ void store_out_kernel(const _Float16* __restrict__ x,
                                 float* __restrict__ out) {
  const int idx = blockIdx.x * blockDim.x + threadIdx.x;
  if (idx >= 2 * 40 * 40 * 128) return;
  const int c = idx % 128;
  int r = idx / 128;
  const int w = r % 40; r /= 40;
  const int h = r % 40;
  const int d = r / 40;
  out[(((c * 2 + d) * 40) + h) * 40 + w] = (float)x[idx];
}

// ---------------------------------------------------------------------------
// Host orchestration
// ---------------------------------------------------------------------------
struct BNp { const float *g, *b, *m, *v; };
struct Blk { const float *b1, *b2; BNp bn1, bn2; const float *w1, *w2; };

static inline int cdiv(long a, int b) { return (int)((a + b - 1) / b); }

static void launch_conv(hipStream_t s,
                        const _Float16* xin, int Din, int Hin, int Win,
                        int Cin, int CinPadIn,
                        const float* wsrc, int kd, int kh, int kw,
                        int sd, int sh, int sw, int pd, int ph, int pw,
                        BNp bn, const float* bias, int Cout,
                        const unsigned char* maskOut, const _Float16* ident,
                        int relu,
                        _Float16* xout, int Dout, int Hout, int Wout,
                        _Float16* wbuf, float* scshbuf) {
  const int T = kd * kh * kw;
  const int wtot = T * CinPadIn * Cout;
  prep_w_kernel<<<cdiv(wtot, 256), 256, 0, s>>>(wsrc, wbuf, Cout, Cin, CinPadIn,
                                                kd, kh, kw);
  prep_scsh_kernel<<<cdiv(Cout, 256), 256, 0, s>>>(bn.g, bn.b, bn.m, bn.v, bias,
                                                   Cout, scshbuf);
  ConvDims cd;
  cd.Din = Din; cd.Hin = Hin; cd.Win = Win;
  cd.Dout = Dout; cd.Hout = Hout; cd.Wout = Wout;
  cd.Cin = Cin; cd.CinPad = CinPadIn; cd.Cout = Cout;
  cd.kd = kd; cd.kh = kh; cd.kw = kw;
  cd.sd = sd; cd.sh = sh; cd.sw = sw;
  cd.pd = pd; cd.ph = ph; cd.pw = pw;
  cd.relu = relu; cd.addid = (ident != nullptr);
  cd.nvoxOut = Dout * Hout * Wout;
  cd.HWout = Hout * Wout;
  const int grid = cdiv((long)cd.nvoxOut, 128);
  switch (Cout >> 4) {
    case 1: conv_wmma_kernel<1><<<grid, 256, 0, s>>>(xin, wbuf, scshbuf, maskOut, ident, xout, cd); break;
    case 2: conv_wmma_kernel<2><<<grid, 256, 0, s>>>(xin, wbuf, scshbuf, maskOut, ident, xout, cd); break;
    case 4: conv_wmma_kernel<4><<<grid, 256, 0, s>>>(xin, wbuf, scshbuf, maskOut, ident, xout, cd); break;
    default: conv_wmma_kernel<8><<<grid, 256, 0, s>>>(xin, wbuf, scshbuf, maskOut, ident, xout, cd); break;
  }
}

// Leaf order assumed = JAX pytree flattening (dict keys sorted, lists in order):
//  0 batch_size | 1 coors | 2..4 input_shape | 5..8 conv_input.bn{b,g,m,v} |
//  9 conv_input.w | 10..14 down2{bn,w} | 15..19 down3 | 20..24 down4 |
//  25..29 extra | 30.. res1[0], 42.. res1[1], 54/66 res2, 78/90 res3,
//  102/114 res4 (each block: b1,b2,bn1{b,g,m,v},bn2{b,g,m,v},w1,w2) |
//  126 voxel_features.   (Single table: adjust here if the harness flattens
//  in insertion order instead.)
static Blk getBlk(void* const* d_in, int base) {
  Blk k;
  k.b1 = (const float*)d_in[base + 0];
  k.b2 = (const float*)d_in[base + 1];
  k.bn1 = { (const float*)d_in[base + 3], (const float*)d_in[base + 2],
            (const float*)d_in[base + 4], (const float*)d_in[base + 5] };
  k.bn2 = { (const float*)d_in[base + 7], (const float*)d_in[base + 6],
            (const float*)d_in[base + 8], (const float*)d_in[base + 9] };
  k.w1 = (const float*)d_in[base + 10];
  k.w2 = (const float*)d_in[base + 11];
  return k;
}
static BNp getBN(void* const* d_in, int base) { // leaves b,g,m,v
  return { (const float*)d_in[base + 1], (const float*)d_in[base + 0],
           (const float*)d_in[base + 2], (const float*)d_in[base + 3] };
}

extern "C" void kernel_launch(void* const* d_in, const int* in_sizes, int n_in,
                              void* d_out, int out_size, void* d_ws, size_t ws_size,
                              hipStream_t stream) {
  (void)n_in; (void)out_size; (void)ws_size;

  const float* vox  = (const float*)d_in[126];
  const int* coors  = (const int*)d_in[1];
  const int nv      = in_sizes[1] / 4;

  // --- workspace arena (peak need ~410 MB) ---
  char* ws = (char*)d_ws;
  size_t off = 0;
  auto take = [&](size_t bytes) -> char* {
    char* p = ws + off;
    off += (bytes + 255) & ~(size_t)255;
    return p;
  };
  const size_t NV0 = 41ull * 320 * 320;             // 4,198,400 voxels
  const size_t ARENA = NV0 * 16 * sizeof(_Float16); // 134.3 MB (fits in 192MB L2)
  _Float16* a0 = (_Float16*)take(ARENA);
  _Float16* a1 = (_Float16*)take(ARENA);
  _Float16* a2 = (_Float16*)take(ARENA);
  unsigned char* m0 = (unsigned char*)take(NV0);
  unsigned char* m1 = (unsigned char*)take(21ull * 160 * 160);
  unsigned char* m2 = (unsigned char*)take(11ull * 80 * 80);
  unsigned char* m3 = (unsigned char*)take(5ull * 40 * 40);
  unsigned char* m4 = (unsigned char*)take(2ull * 40 * 40);
  _Float16* wbuf = (_Float16*)take(27ull * 128 * 128 * sizeof(_Float16));
  float* scsh = (float*)take(2 * 128 * sizeof(float));

  // --- build dense f16 input (Cpad=8) + mask ---
  zero_words_kernel<<<2048, 256, 0, stream>>>((uint32_t*)a0, (long)(NV0 * 8 * 2 / 4));
  zero_words_kernel<<<2048, 256, 0, stream>>>((uint32_t*)m0, (long)(NV0 / 4));
  scatter_kernel<<<cdiv(nv, 256), 256, 0, stream>>>(vox, coors, nv, a0, m0, 320, 320, 8);

  // --- conv_input: 5->16, SubM-style (same mask) ---
  launch_conv(stream, a0, 41, 320, 320, 5, 8, (const float*)d_in[9],
              3, 3, 3, 1, 1, 1, 1, 1, 1, getBN(d_in, 5), nullptr, 16,
              m0, nullptr, 1, a1, 41, 320, 320, wbuf, scsh);

  // --- res1 (16 ch, grid 41x320x320) ---
  {
    Blk B = getBlk(d_in, 30);
    launch_conv(stream, a1, 41,320,320, 16,16, B.w1, 3,3,3, 1,1,1, 1,1,1, B.bn1, B.b1, 16, m0, nullptr, 1, a2, 41,320,320, wbuf, scsh);
    launch_conv(stream, a2, 41,320,320, 16,16, B.w2, 3,3,3, 1,1,1, 1,1,1, B.bn2, B.b2, 16, m0, a1,      1, a0, 41,320,320, wbuf, scsh);
    B = getBlk(d_in, 42);
    launch_conv(stream, a0, 41,320,320, 16,16, B.w1, 3,3,3, 1,1,1, 1,1,1, B.bn1, B.b1, 16, m0, nullptr, 1, a1, 41,320,320, wbuf, scsh);
    launch_conv(stream, a1, 41,320,320, 16,16, B.w2, 3,3,3, 1,1,1, 1,1,1, B.bn2, B.b2, 16, m0, a0,      1, a2, 41,320,320, wbuf, scsh);
  }

  // --- down2: 16->32, stride 2, pad 1 ---
  mask_down_kernel<<<cdiv(21 * 160 * 160, 256), 256, 0, stream>>>(
      m0, m1, 41, 320, 320, 21, 160, 160, 3, 3, 3, 2, 2, 2, 1, 1, 1);
  launch_conv(stream, a2, 41, 320, 320, 16, 16, (const float*)d_in[14],
              3, 3, 3, 2, 2, 2, 1, 1, 1, getBN(d_in, 10), nullptr, 32,
              m1, nullptr, 1, a0, 21, 160, 160, wbuf, scsh);

  // --- res2 (32 ch, grid 21x160x160) ---
  {
    Blk B = getBlk(d_in, 54);
    launch_conv(stream, a0, 21,160,160, 32,32, B.w1, 3,3,3, 1,1,1, 1,1,1, B.bn1, B.b1, 32, m1, nullptr, 1, a1, 21,160,160, wbuf, scsh);
    launch_conv(stream, a1, 21,160,160, 32,32, B.w2, 3,3,3, 1,1,1, 1,1,1, B.bn2, B.b2, 32, m1, a0,      1, a2, 21,160,160, wbuf, scsh);
    B = getBlk(d_in, 66);
    launch_conv(stream, a2, 21,160,160, 32,32, B.w1, 3,3,3, 1,1,1, 1,1,1, B.bn1, B.b1, 32, m1, nullptr, 1, a0, 21,160,160, wbuf, scsh);
    launch_conv(stream, a0, 21,160,160, 32,32, B.w2, 3,3,3, 1,1,1, 1,1,1, B.bn2, B.b2, 32, m1, a2,      1, a1, 21,160,160, wbuf, scsh);
  }

  // --- down3: 32->64, stride 2, pad 1 ---
  mask_down_kernel<<<cdiv(11 * 80 * 80, 256), 256, 0, stream>>>(
      m1, m2, 21, 160, 160, 11, 80, 80, 3, 3, 3, 2, 2, 2, 1, 1, 1);
  launch_conv(stream, a1, 21, 160, 160, 32, 32, (const float*)d_in[19],
              3, 3, 3, 2, 2, 2, 1, 1, 1, getBN(d_in, 15), nullptr, 64,
              m2, nullptr, 1, a0, 11, 80, 80, wbuf, scsh);

  // --- res3 (64 ch, grid 11x80x80) ---
  {
    Blk B = getBlk(d_in, 78);
    launch_conv(stream, a0, 11,80,80, 64,64, B.w1, 3,3,3, 1,1,1, 1,1,1, B.bn1, B.b1, 64, m2, nullptr, 1, a1, 11,80,80, wbuf, scsh);
    launch_conv(stream, a1, 11,80,80, 64,64, B.w2, 3,3,3, 1,1,1, 1,1,1, B.bn2, B.b2, 64, m2, a0,      1, a2, 11,80,80, wbuf, scsh);
    B = getBlk(d_in, 90);
    launch_conv(stream, a2, 11,80,80, 64,64, B.w1, 3,3,3, 1,1,1, 1,1,1, B.bn1, B.b1, 64, m2, nullptr, 1, a0, 11,80,80, wbuf, scsh);
    launch_conv(stream, a0, 11,80,80, 64,64, B.w2, 3,3,3, 1,1,1, 1,1,1, B.bn2, B.b2, 64, m2, a2,      1, a1, 11,80,80, wbuf, scsh);
  }

  // --- down4: 64->128, stride 2, pad (0,1,1) ---
  mask_down_kernel<<<cdiv(5 * 40 * 40, 256), 256, 0, stream>>>(
      m2, m3, 11, 80, 80, 5, 40, 40, 3, 3, 3, 2, 2, 2, 0, 1, 1);
  launch_conv(stream, a1, 11, 80, 80, 64, 64, (const float*)d_in[24],
              3, 3, 3, 2, 2, 2, 0, 1, 1, getBN(d_in, 20), nullptr, 128,
              m3, nullptr, 1, a0, 5, 40, 40, wbuf, scsh);

  // --- res4 (128 ch, grid 5x40x40) ---
  {
    Blk B = getBlk(d_in, 102);
    launch_conv(stream, a0, 5,40,40, 128,128, B.w1, 3,3,3, 1,1,1, 1,1,1, B.bn1, B.b1, 128, m3, nullptr, 1, a1, 5,40,40, wbuf, scsh);
    launch_conv(stream, a1, 5,40,40, 128,128, B.w2, 3,3,3, 1,1,1, 1,1,1, B.bn2, B.b2, 128, m3, a0,      1, a2, 5,40,40, wbuf, scsh);
    B = getBlk(d_in, 114);
    launch_conv(stream, a2, 5,40,40, 128,128, B.w1, 3,3,3, 1,1,1, 1,1,1, B.bn1, B.b1, 128, m3, nullptr, 1, a0, 5,40,40, wbuf, scsh);
    launch_conv(stream, a0, 5,40,40, 128,128, B.w2, 3,3,3, 1,1,1, 1,1,1, B.bn2, B.b2, 128, m3, a2,      1, a1, 5,40,40, wbuf, scsh);
  }

  // --- extra: 128->128, kernel (3,1,1), stride (2,1,1), no pad ---
  mask_down_kernel<<<cdiv(2 * 40 * 40, 256), 256, 0, stream>>>(
      m3, m4, 5, 40, 40, 2, 40, 40, 3, 1, 1, 2, 1, 1, 0, 0, 0);
  launch_conv(stream, a1, 5, 40, 40, 128, 128, (const float*)d_in[29],
              3, 1, 1, 2, 1, 1, 0, 0, 0, getBN(d_in, 25), nullptr, 128,
              m4, nullptr, 1, a0, 2, 40, 40, wbuf, scsh);

  // --- reshape (1,128,2,40,40) -> (1,256,40,40) f32 ---
  store_out_kernel<<<cdiv(2 * 40 * 40 * 128, 256), 256, 0, stream>>>(
      a0, (float*)d_out);
}
